// MultiScaleSpringGNN_37082747634274
// MI455X (gfx1250) — matrix operand order repaired and checked by reading, not compiled, not run on hardware
//
#include <hip/hip_runtime.h>

// ---------------------------------------------------------------------------
// MI455X (gfx1250) implementation of MultiScaleSpringGNN forward.
// All dense GEMMs + the N^2 multi-head attention run on v_wmma_f32_16x16x32_bf16
// (wave32, fp32 accumulate). Each GEMM wave computes a 16x64 output strip
// (4 WMMAs per k-step sharing one A-fragment) to raise arithmetic intensity.
// Irregular parts (GCN scatter, top-k, k=32 gather attention, tiny-N linears)
// are scalar VALU kernels.
// ---------------------------------------------------------------------------

static constexpr int NNODE = 6000;
static constexpr int NEDGE = 192000;
static constexpr int INDIM = 130;   // 128 feats + 2 coords
static constexpr int HDIM  = 256;
static constexpr int KTOP  = 32;

typedef __bf16 bf16x16 __attribute__((ext_vector_type(16)));
typedef float  f32x8   __attribute__((ext_vector_type(8)));

static __device__ __forceinline__ f32x8 wmma_bf16(bf16x16 a, bf16x16 b, f32x8 c) {
  // D = A(16x32) * B(32x16) + C, fp32 accumulate
  return __builtin_amdgcn_wmma_f32_16x16x32_bf16(false, a, false, b, (short)0, c,
                                                 false, false);
}

// ---------------------------------------------------------------------------
// WMMA GEMM: C[M,N] = A[M,K] @ B (+bias, +relu).  One wave -> 16x64 strip.
// TRANSB=0: B is (K,N) row-major (ldb=N).  TRANSB=1: B is (N,K) row-major
// (ldb=K), i.e. computes A @ B^T.  N must be a multiple of 64, K of 32,
// M of 16.  A/B frags use contiguous k = 16*half + e; C layout per CDNA5 ISA 7.12.2.
// ---------------------------------------------------------------------------
template <bool TRANSB, bool RELU>
__global__ __launch_bounds__(32)
void k_gemm(const float* __restrict__ A, int lda,
            const float* __restrict__ B, int ldb,
            const float* __restrict__ bias,
            float* __restrict__ C, int ldc, int ocol, int K)
{
  const int lane = threadIdx.x;
  const int half = lane >> 4;
  const int l16  = lane & 15;
  const int ib   = blockIdx.x * 16;
  const int jb   = blockIdx.y * 64;
  const int row  = ib + l16;

  f32x8 acc[4];
  const f32x8 vzero = {};
#pragma unroll
  for (int ct = 0; ct < 4; ++ct) acc[ct] = vzero;

  for (int k0 = 0; k0 < K; k0 += 32) {
    // A: lane<16 -> K 0..15; lane>=16 -> K 16..31
    bf16x16 a;
    const float* ap = A + (size_t)row * lda + k0 + 16 * half;
#pragma unroll
    for (int e = 0; e < 16; ++e) a[e] = (__bf16)ap[e];
    // B: lane ell holds column (ell&15), K = 16*half + e
#pragma unroll
    for (int ct = 0; ct < 4; ++ct) {
      const int col = jb + ct * 16 + l16;
      bf16x16 b;
      if (TRANSB) {
        const float* bp = B + (size_t)col * ldb + k0 + 16 * half;
#pragma unroll
        for (int e = 0; e < 16; ++e) b[e] = (__bf16)bp[e];
      } else {
        const float* bp = B + (size_t)(k0 + 16 * half) * ldb + col;
#pragma unroll
        for (int e = 0; e < 16; ++e) b[e] = (__bf16)bp[(size_t)e * ldb];
      }
      acc[ct] = wmma_bf16(a, b, acc[ct]);
    }
  }
#pragma unroll
  for (int ct = 0; ct < 4; ++ct) {
    const int col = jb + ct * 16 + l16;
    const float bb = bias ? bias[col] : 0.0f;
#pragma unroll
    for (int r = 0; r < 8; ++r) {
      float v = acc[ct][r] + bb;
      if (RELU) v = fmaxf(v, 0.0f);
      C[(size_t)(ib + r + 8 * half) * ldc + ocol + col] = v;
    }
  }
}

// ---------------------------------------------------------------------------
// Small/odd-shape linear: out[n, ocol+o] = dot(A[n, acol:acol+K], W[:,o]) + b[o]
// W is (K, Nout) row-major. One thread per output element.
// ---------------------------------------------------------------------------
template <bool RELU>
__global__ void k_lin_small(const float* __restrict__ A, int lda, int acol,
                            const float* __restrict__ W,
                            const float* __restrict__ bias,
                            float* __restrict__ out, int ldo, int ocol,
                            int M, int K, int Nout)
{
  const int t = blockIdx.x * 256 + threadIdx.x;
  if (t >= M * Nout) return;
  const int n = t / Nout, o = t % Nout;
  const float* a = A + (size_t)n * lda + acol;
  float acc = bias ? bias[o] : 0.0f;
  for (int k = 0; k < K; ++k) acc += a[k] * W[(size_t)k * Nout + o];
  if (RELU) acc = fmaxf(acc, 0.0f);
  out[(size_t)n * ldo + ocol + o] = acc;
}

// ---------------------------------------------------------------------------
// GCN support: degree (in-degree; +1 self-loop added in k_dinv), dinv,
// edge scatter with symmetric norm, self-loop + bias + relu finish.
// ---------------------------------------------------------------------------
__global__ void k_deg(const int* __restrict__ ei, float* __restrict__ deg)
{
  const int e = blockIdx.x * 256 + threadIdx.x;
  if (e < NEDGE) atomicAdd(&deg[ei[NEDGE + e]], 1.0f);
}

__global__ void k_dinv(const float* __restrict__ deg, float* __restrict__ dinv)
{
  const int n = blockIdx.x * 256 + threadIdx.x;
  if (n < NNODE) dinv[n] = rsqrtf(deg[n] + 1.0f);  // +1 = self loop
}

__global__ void k_scatter(const int* __restrict__ ei,
                          const float* __restrict__ dinv,
                          const float* __restrict__ hw,
                          float* __restrict__ out)
{
  const int e = blockIdx.x;
  const int c = threadIdx.x;
  __shared__ float wsh;
  __shared__ int ssh, dsh;
  if (c == 0) {
    const int s = ei[e], d = ei[NEDGE + e];
    ssh = s; dsh = d;
    wsh = dinv[s] * dinv[d];
  }
  __syncthreads();
  atomicAdd(&out[(size_t)dsh * HDIM + c], wsh * hw[(size_t)ssh * HDIM + c]);
}

__global__ void k_gcn_finish(const float* __restrict__ dinv,
                             const float* __restrict__ hw,
                             const float* __restrict__ bias,
                             float* __restrict__ out)
{
  const int t = blockIdx.x * 256 + threadIdx.x;  // grid = NNODE blocks
  const int n = t >> 8, c = t & 255;
  const float di = dinv[n];
  const float v = out[t] + di * di * hw[t] + bias[c];
  out[t] = fmaxf(v, 0.0f);
}

// ---------------------------------------------------------------------------
// Flash-style full MHA: heads=4, dh=64, streaming softmax over key chunks of
// 32.  One wave per (16-query block, head). qkv is (N, 768) = [q|k|v].
// Output (pre out-projection) written to out (N, 256).
// ---------------------------------------------------------------------------
__global__ __launch_bounds__(32)
void k_flash(const float* __restrict__ qkv, float* __restrict__ out)
{
  const int lane = threadIdx.x;
  const int half = lane >> 4;
  const int l16  = lane & 15;
  const int qb   = blockIdx.x;   // 0..374
  const int head = blockIdx.y;   // 0..3
  const int qrow = qb * 16 + l16;

  __shared__ float ps[32][17];   // P chunk: [key within chunk][query row]

  // Q A-frags for the two 32-wide K-dim slices of dh=64
  bf16x16 aq[2];
  {
    const float* qp = qkv + (size_t)qrow * 768 + head * 64;
#pragma unroll
    for (int t = 0; t < 2; ++t) {
      const float* p = qp + t * 32 + 16 * half;
#pragma unroll
      for (int e = 0; e < 16; ++e) aq[t][e] = (__bf16)p[e];
    }
  }

  const f32x8 vzero = {};
  f32x8 oc[4];
#pragma unroll
  for (int i = 0; i < 4; ++i) oc[i] = vzero;
  float m_run[8], l_run[8];
#pragma unroll
  for (int r = 0; r < 8; ++r) { m_run[r] = -3.0e38f; l_run[r] = 0.0f; }

  for (int kb = 0; kb < NNODE; kb += 32) {
    // ---- S = Q K^T for 32 keys (two 16-wide tiles) ----
    f32x8 s[2];
#pragma unroll
    for (int sub = 0; sub < 2; ++sub) {
      const int key  = kb + sub * 16 + l16;
      const int keyc = key < NNODE ? key : NNODE - 1;
      f32x8 acc = vzero;
#pragma unroll
      for (int t = 0; t < 2; ++t) {
        bf16x16 bk;
        const float* kp = qkv + (size_t)keyc * 768 + 256 + head * 64 + t * 32 + 16 * half;
#pragma unroll
        for (int e = 0; e < 16; ++e) bk[e] = (__bf16)kp[e];
        acc = wmma_bf16(aq[t], bk, acc);
      }
      s[sub] = acc;
    }
    // scale (1/sqrt(64)) + mask out-of-range key columns (per-lane column)
    const bool v1 = (kb + 16 + l16) < NNODE;
    float mx[8];
#pragma unroll
    for (int r = 0; r < 8; ++r) {
      const float a0 = s[0][r] * 0.125f;
      const float a1 = v1 ? s[1][r] * 0.125f : -3.0e38f;
      s[0][r] = a0; s[1][r] = a1;
      mx[r] = fmaxf(a0, a1);
    }
    // row reductions stay within 16-lane halves (xor masks 1,2,4,8)
#pragma unroll
    for (int d = 1; d < 16; d <<= 1)
#pragma unroll
      for (int r = 0; r < 8; ++r) mx[r] = fmaxf(mx[r], __shfl_xor(mx[r], d, 32));
    float alpha[8];
#pragma unroll
    for (int r = 0; r < 8; ++r) {
      const float mnew = fmaxf(m_run[r], mx[r]);
      alpha[r] = __expf(m_run[r] - mnew);
      m_run[r] = mnew;
    }
    float lc[8];
#pragma unroll
    for (int r = 0; r < 8; ++r) {
      const float p0 = __expf(s[0][r] - m_run[r]);
      const float p1 = __expf(s[1][r] - m_run[r]);
      ps[l16][r + 8 * half]      = p0;   // C-layout -> [key][row] in LDS
      ps[16 + l16][r + 8 * half] = p1;
      lc[r] = p0 + p1;
    }
#pragma unroll
    for (int d = 1; d < 16; d <<= 1)
#pragma unroll
      for (int r = 0; r < 8; ++r) lc[r] += __shfl_xor(lc[r], d, 32);
#pragma unroll
    for (int r = 0; r < 8; ++r) l_run[r] = l_run[r] * alpha[r] + lc[r];
    __syncthreads();

    // ---- rebuild P in A-matrix layout, accumulate O += P @ V ----
    bf16x16 ap;
#pragma unroll
    for (int e = 0; e < 16; ++e) {
      const int kk = e + 16 * half;
      ap[e] = (__bf16)ps[kk][l16];
    }
#pragma unroll
    for (int ct = 0; ct < 4; ++ct) {
      bf16x16 bv;
      const int colv = 512 + head * 64 + ct * 16 + l16;
#pragma unroll
      for (int e = 0; e < 16; ++e) {
        const int key  = kb + e + 16 * half;
        const int keyc = key < NNODE ? key : NNODE - 1;  // P==0 for invalid keys
        bv[e] = (__bf16)qkv[(size_t)keyc * 768 + colv];
      }
#pragma unroll
      for (int r = 0; r < 8; ++r) oc[ct][r] *= alpha[r];
      oc[ct] = wmma_bf16(ap, bv, oc[ct]);
    }
    __syncthreads();
  }

  float inv[8];
#pragma unroll
  for (int r = 0; r < 8; ++r) inv[r] = 1.0f / l_run[r];
#pragma unroll
  for (int ct = 0; ct < 4; ++ct)
#pragma unroll
    for (int r = 0; r < 8; ++r)
      out[(size_t)(qb * 16 + r + 8 * half) * HDIM + head * 64 + ct * 16 + l16] =
          oc[ct][r] * inv[r];
}

// ---------------------------------------------------------------------------
// Top-32 nearest neighbors by coordinate distance (includes self).
// One thread per query node; candidate coords staged in LDS.
// ---------------------------------------------------------------------------
__global__ void k_topk(const float* __restrict__ x,
                       float* __restrict__ tkd, int* __restrict__ tki)
{
  __shared__ float sc[128][2];
  const int n = blockIdx.x * 128 + threadIdx.x;
  float cx = 0.0f, cy = 0.0f;
  if (n < NNODE) { cx = x[(size_t)n * INDIM + 128]; cy = x[(size_t)n * INDIM + 129]; }

  float bd[KTOP]; int bi[KTOP];
#pragma unroll
  for (int t = 0; t < KTOP; ++t) { bd[t] = 3.0e38f; bi[t] = 0; }

  for (int base = 0; base < NNODE; base += 128) {
    __syncthreads();
    const int j = base + threadIdx.x;
    if (j < NNODE) {
      sc[threadIdx.x][0] = x[(size_t)j * INDIM + 128];
      sc[threadIdx.x][1] = x[(size_t)j * INDIM + 129];
    }
    __syncthreads();
    const int lim = (NNODE - base) < 128 ? (NNODE - base) : 128;
    for (int t = 0; t < lim; ++t) {
      const float dx = cx - sc[t][0];
      const float dy = cy - sc[t][1];
      const float d  = sqrtf(fmaxf(dx * dx + dy * dy, 0.0f) + 1e-12f);
      if (d < bd[KTOP - 1]) {
        int p = KTOP - 1;
        while (p > 0 && bd[p - 1] > d) { bd[p] = bd[p - 1]; bi[p] = bi[p - 1]; --p; }
        bd[p] = d; bi[p] = base + t;
      }
    }
  }
  if (n < NNODE) {
    for (int t = 0; t < KTOP; ++t) {
      tkd[(size_t)n * KTOP + t] = bd[t];
      tki[(size_t)n * KTOP + t] = bi[t];
    }
  }
}

// ---------------------------------------------------------------------------
// Global sparse attention over the 32 neighbors: one wave per node.
// Lane k scores neighbor k (256-dot), wave softmax, then each lane
// accumulates 8 output channels.
// ---------------------------------------------------------------------------
__global__ __launch_bounds__(32)
void k_gather(const float* __restrict__ Q, const float* __restrict__ K,
              const float* __restrict__ V, const int* __restrict__ tki,
              const float* __restrict__ tkd, float* __restrict__ gfeat)
{
  const int n = blockIdx.x;
  const int lane = threadIdx.x;
  __shared__ float sa[KTOP];
  __shared__ int   si[KTOP];

  const int   id = tki[(size_t)n * KTOP + lane];
  const float dk = tkd[(size_t)n * KTOP + lane];
  const float* qp = Q + (size_t)n * HDIM;
  const float* kp = K + (size_t)id * HDIM;
  float s = 0.0f;
  for (int c = 0; c < HDIM; ++c) s += qp[c] * kp[c];
  s = s * 0.0625f + 1.0f / (dk + 1e-6f);   // 1/sqrt(256) + inverse-distance bias

  float mx = s;
#pragma unroll
  for (int d = 1; d < 32; d <<= 1) mx = fmaxf(mx, __shfl_xor(mx, d, 32));
  const float p = __expf(s - mx);
  float sum = p;
#pragma unroll
  for (int d = 1; d < 32; d <<= 1) sum += __shfl_xor(sum, d, 32);
  sa[lane] = p / sum;
  si[lane] = id;
  __syncthreads();

  float acc[8] = {0, 0, 0, 0, 0, 0, 0, 0};
  const int c0 = lane * 8;
  for (int k = 0; k < KTOP; ++k) {
    const float w = sa[k];
    const float* vp = V + (size_t)si[k] * HDIM + c0;
#pragma unroll
    for (int u = 0; u < 8; ++u) acc[u] += w * vp[u];
  }
#pragma unroll
  for (int u = 0; u < 8; ++u) gfeat[(size_t)n * HDIM + c0 + u] = acc[u];
}

// ---------------------------------------------------------------------------
// Combine-branch input: f6[n] = [w0*f_l, w1*f_m, w2*f_g], w = softmax(w3)
// ---------------------------------------------------------------------------
__global__ void k_f6(const float* __restrict__ w3,
                     const float* __restrict__ fl, const float* __restrict__ fm,
                     const float* __restrict__ fg, float* __restrict__ f6)
{
  const int n = blockIdx.x * 256 + threadIdx.x;
  if (n >= NNODE) return;
  const float a = w3[0], b = w3[1], c = w3[2];
  const float mx = fmaxf(a, fmaxf(b, c));
  const float ea = __expf(a - mx), eb = __expf(b - mx), ec = __expf(c - mx);
  const float inv = 1.0f / (ea + eb + ec);
  const float w0 = ea * inv, w1 = eb * inv, w2 = ec * inv;
  f6[(size_t)n * 6 + 0] = w0 * fl[n * 2];
  f6[(size_t)n * 6 + 1] = w0 * fl[n * 2 + 1];
  f6[(size_t)n * 6 + 2] = w1 * fm[n * 2];
  f6[(size_t)n * 6 + 3] = w1 * fm[n * 2 + 1];
  f6[(size_t)n * 6 + 4] = w2 * fg[n * 2];
  f6[(size_t)n * 6 + 5] = w2 * fg[n * 2 + 1];
}

// ---------------------------------------------------------------------------
// Host-side orchestration
// ---------------------------------------------------------------------------
extern "C" void kernel_launch(void* const* d_in, const int* in_sizes, int n_in,
                              void* d_out, int out_size, void* d_ws, size_t ws_size,
                              hipStream_t stream)
{
  (void)in_sizes; (void)n_in; (void)out_size; (void)ws_size;

  // Flat leaf order (setup_inputs() insertion order):
  // 0:x 1:edge_index
  // local : 2 node.w 3 node.b 4 coord.w 5 coord.b 6 gcn.w 7 gcn.b 8 head.w 9 head.b
  // medium: 10 node.w 11 node.b 12 coord.w 13 coord.b 14 gcn1.w 15 gcn1.b
  //         16 gcn2.w 17 gcn2.b 18 gcn3.w 19 gcn3.b 20 attn.in_w 21 attn.in_b
  //         22 attn.out_w 23 attn.out_b 24 head.w 25 head.b
  // global: 26 node.w 27 node.b 28 coord.w 29 coord.b 30 q.w 31 q.b 32 k.w 33 k.b
  //         34 v.w 35 v.b 36 head.w 37 head.b
  // comb  : 38 w3 39 l1.w 40 l1.b 41 l2.w 42 l2.b 43 l3.w 44 l3.b
  const float* x  = (const float*)d_in[0];
  const int*   ei = (const int*)d_in[1];
  const float* P[45];
  for (int i = 0; i < 45; ++i) P[i] = (const float*)d_in[i];

  // Workspace layout (floats)
  float* ws   = (float*)d_ws;
  float* hc   = ws;                               // 6000 x 320
  float* hw   = hc + (size_t)NNODE * 320;         // 6000 x 256
  float* hA   = hw + (size_t)NNODE * HDIM;        // 6000 x 256
  float* hB   = hA + (size_t)NNODE * HDIM;        // 6000 x 256
  float* qkvb = hB + (size_t)NNODE * HDIM;        // 6000 x 768
  float* deg  = qkvb + (size_t)NNODE * 768;       // 6000
  float* dinv = deg + NNODE;                      // 6000
  float* fl   = dinv + NNODE;                     // 6000 x 2
  float* fm   = fl + NNODE * 2;
  float* fg   = fm + NNODE * 2;
  float* f6   = fg + NNODE * 2;                   // 6000 x 6
  float* tkd  = f6 + NNODE * 6;                   // 6000 x 32
  int*   tki  = (int*)(tkd + NNODE * KTOP);       // 6000 x 32 ints

  const dim3 gN256(NNODE / 16, HDIM / 64);        // (375, 4)
  const dim3 gN768(NNODE / 16, 768 / 64);         // (375, 12)
  const dim3 gN128(NNODE / 16, 128 / 64);         // (375, 2)
  const size_t hbytes = (size_t)NNODE * HDIM * sizeof(float);

  // ---- degree / dinv (shared by all GCN layers) ----
  hipMemsetAsync(deg, 0, NNODE * sizeof(float), stream);
  k_deg<<<(NEDGE + 255) / 256, 256, 0, stream>>>(ei, deg);
  k_dinv<<<(NNODE + 255) / 256, 256, 0, stream>>>(deg, dinv);

  // ================= local branch =================
  k_gemm<false, false><<<gN256, 32, 0, stream>>>(x, INDIM, P[2], 256, P[3], hc, 320, 0, 128);
  k_lin_small<false><<<(NNODE * 64 + 255) / 256, 256, 0, stream>>>(x, INDIM, 128, P[4], P[5], hc, 320, 256, NNODE, 2, 64);
  k_gemm<false, false><<<gN256, 32, 0, stream>>>(hc, 320, P[6], 256, nullptr, hw, 256, 0, 320);
  hipMemsetAsync(hA, 0, hbytes, stream);
  k_scatter<<<NEDGE, 256, 0, stream>>>(ei, dinv, hw, hA);
  k_gcn_finish<<<NNODE, 256, 0, stream>>>(dinv, hw, P[7], hA);
  k_lin_small<false><<<(NNODE * 2 + 255) / 256, 256, 0, stream>>>(hA, 256, 0, P[8], P[9], fl, 2, 0, NNODE, 256, 2);

  // ================= medium branch =================
  k_gemm<false, false><<<gN256, 32, 0, stream>>>(x, INDIM, P[10], 256, P[11], hc, 320, 0, 128);
  k_lin_small<false><<<(NNODE * 64 + 255) / 256, 256, 0, stream>>>(x, INDIM, 128, P[12], P[13], hc, 320, 256, NNODE, 2, 64);
  // gcn1
  k_gemm<false, false><<<gN256, 32, 0, stream>>>(hc, 320, P[14], 256, nullptr, hw, 256, 0, 320);
  hipMemsetAsync(hA, 0, hbytes, stream);
  k_scatter<<<NEDGE, 256, 0, stream>>>(ei, dinv, hw, hA);
  k_gcn_finish<<<NNODE, 256, 0, stream>>>(dinv, hw, P[15], hA);
  // gcn2
  k_gemm<false, false><<<gN256, 32, 0, stream>>>(hA, 256, P[16], 256, nullptr, hw, 256, 0, 256);
  hipMemsetAsync(hB, 0, hbytes, stream);
  k_scatter<<<NEDGE, 256, 0, stream>>>(ei, dinv, hw, hB);
  k_gcn_finish<<<NNODE, 256, 0, stream>>>(dinv, hw, P[17], hB);
  // gcn3
  k_gemm<false, false><<<gN256, 32, 0, stream>>>(hB, 256, P[18], 256, nullptr, hw, 256, 0, 256);
  hipMemsetAsync(hA, 0, hbytes, stream);
  k_scatter<<<NEDGE, 256, 0, stream>>>(ei, dinv, hw, hA);
  k_gcn_finish<<<NNODE, 256, 0, stream>>>(dinv, hw, P[19], hA);
  // MHA: qkv = h @ in_w^T + in_b ; flash ; out = o @ out_w^T + out_b
  k_gemm<true, false><<<gN768, 32, 0, stream>>>(hA, 256, P[20], 256, P[21], qkvb, 768, 0, 256);
  k_flash<<<dim3(NNODE / 16, 4), 32, 0, stream>>>(qkvb, hB);
  k_gemm<true, false><<<gN256, 32, 0, stream>>>(hB, 256, P[22], 256, P[23], hw, 256, 0, 256);
  k_lin_small<false><<<(NNODE * 2 + 255) / 256, 256, 0, stream>>>(hw, 256, 0, P[24], P[25], fm, 2, 0, NNODE, 256, 2);

  // ================= global branch =================
  k_gemm<false, false><<<gN256, 32, 0, stream>>>(x, INDIM, P[26], 256, P[27], hc, 320, 0, 128);
  k_lin_small<false><<<(NNODE * 64 + 255) / 256, 256, 0, stream>>>(x, INDIM, 128, P[28], P[29], hc, 320, 256, NNODE, 2, 64);
  k_gemm<false, false><<<gN256, 32, 0, stream>>>(hc, 320, P[30], 256, P[31], hA, 256, 0, 320);  // Q
  k_gemm<false, false><<<gN256, 32, 0, stream>>>(hc, 320, P[32], 256, P[33], hB, 256, 0, 320);  // K
  k_gemm<false, false><<<gN256, 32, 0, stream>>>(hc, 320, P[34], 256, P[35], hw, 256, 0, 320);  // V
  k_topk<<<(NNODE + 127) / 128, 128, 0, stream>>>(x, tkd, tki);
  k_gather<<<NNODE, 32, 0, stream>>>(hA, hB, hw, tki, tkd, hc);  // gfeat -> hc (ld 256)
  k_lin_small<false><<<(NNODE * 2 + 255) / 256, 256, 0, stream>>>(hc, 256, 0, P[36], P[37], fg, 2, 0, NNODE, 256, 2);

  // ================= combine =================
  k_f6<<<(NNODE + 255) / 256, 256, 0, stream>>>(P[38], fl, fm, fg, f6);
  float* h1 = qkvb;                         // 6000 x 256 (reuse)
  float* h2 = qkvb + (size_t)NNODE * HDIM;  // 6000 x 128 (reuse)
  k_lin_small<true><<<(NNODE * 256 + 255) / 256, 256, 0, stream>>>(f6, 6, 0, P[39], P[40], h1, 256, 0, NNODE, 6, 256);
  k_gemm<false, true><<<gN128, 32, 0, stream>>>(h1, 256, P[41], 128, P[42], h2, 128, 0, 256);
  k_lin_small<false><<<(NNODE * 2 + 255) / 256, 256, 0, stream>>>(h2, 128, 0, P[43], P[44], (float*)d_out, 2, 0, NNODE, 128, 2);
}
